// QGNN_7_52243982188640
// MI455X (gfx1250) — matrix operand order repaired and checked by reading, not compiled, not run on hardware
//
#include <hip/hip_runtime.h>

#define N_NODES  65536
#define N_EDGES  524288
#define N_GRAPHS 1024

typedef __attribute__((ext_vector_type(16))) _Float16 v16h;
typedef __attribute__((ext_vector_type(8)))  _Float16 v8h;
typedef __attribute__((ext_vector_type(4)))  _Float16 v4h;
typedef __attribute__((ext_vector_type(8)))  float    v8f;

#define CDIV(a,b) (((a)+(b)-1)/(b))

// ---------------- fragment loaders (CDNA5 WMMA 16x16x32 f16 layouts) -------
// All GEMMs run with K % 32 == 0 (layer 0 is zero-padded 5 -> 32), so the
// loaders are branch-free vector loads.

// A (16x32, f16): lane L<16 -> row L, K = {k0..k0+7, k0+16..k0+23}, k0 = kt*32
//                 lane L>=16 -> row L-16, k0 += 8
static __device__ inline v16h load_a_f16(const _Float16* __restrict__ A, int row,
                                         int K, int kt, int lane) {
  int k0 = kt * 32 + ((lane >> 4) << 3);
  const _Float16* p = A + (size_t)row * K + k0;
  v8h lo = *(const v8h*)(p);
  v8h hi = *(const v8h*)(p + 16);
  v16h a;
#pragma unroll
  for (int i = 0; i < 8; ++i) { a[i] = lo[i]; a[i + 8] = hi[i]; }
  return a;
}

// Same A layout from an f32 source (scatter accumulator), convert on load.
static __device__ inline v16h load_a_f32(const float* __restrict__ A, int row,
                                         int K, int kt, int lane) {
  int k0 = kt * 32 + ((lane >> 4) << 3);
  v8f lo = *(const v8f*)(A + (size_t)row * K + k0);
  v8f hi = *(const v8f*)(A + (size_t)row * K + k0 + 16);
  v16h a;
#pragma unroll
  for (int i = 0; i < 8; ++i) {
    a[i] = (_Float16)lo[i];
    a[i + 8] = (_Float16)hi[i];
  }
  return a;
}

// B (32x16): lane L<16 -> col L, K = {kb..kb+7, kb+16..kb+23} ; lane L>=16 -> col L-16, kb += 8.
// Wt stored transposed [cout, K] f16 -> two 16B loads per lane.
static __device__ inline v16h load_b(const _Float16* __restrict__ Wt, int col,
                                     int K, int kt, int lane) {
  int kb = kt * 32 + ((lane >> 4) << 3);
  const _Float16* p = Wt + (size_t)col * K + kb;
  v8h lo = *(const v8h*)(p);
  v8h hi = *(const v8h*)(p + 16);
  v16h b;
#pragma unroll
  for (int i = 0; i < 8; ++i) { b[i] = lo[i]; b[i + 8] = hi[i]; }
  return b;
}

// ---------------- GEMM: Out = relu( A1@B1^T + A2@B2^T + bias ) -------------
// A1: [M,K] f32 (DUAL path), A2: [M,K] f16, B*: [Ncol,K] f16 transposed.
// Block = 8 waves; each wave owns a 16 x (NT*16) output strip.
// Requires: M % 16 == 0, Ncol % (NT*16) == 0, K % 32 == 0.
template <int NT, bool DUAL>
__global__ __launch_bounds__(256) void gemm_kernel(
    const float* __restrict__ A1, const _Float16* __restrict__ B1,
    const _Float16* __restrict__ A2, const _Float16* __restrict__ B2,
    const float* __restrict__ bias, float* __restrict__ Out,
    int M, int K, int Ncol) {
  int lane = threadIdx.x & 31;
  int wave = threadIdx.x >> 5;
  int row0 = (blockIdx.x * 8 + wave) * 16;
  if (row0 >= M) return;                 // uniform per wave: EXEC stays full
  int nt0 = blockIdx.y * NT;
  int arow = row0 + (lane & 15);
  int colbase = lane & 15;
  v8f acc[NT] = {};
  int ktiles = K >> 5;
  if (DUAL) {
    for (int kt = 0; kt < ktiles; ++kt) {
      v16h a = load_a_f32(A1, arow, K, kt, lane);
#pragma unroll
      for (int t = 0; t < NT; ++t) {
        v16h b = load_b(B1, (nt0 + t) * 16 + colbase, K, kt, lane);
        acc[t] = __builtin_amdgcn_wmma_f32_16x16x32_f16(
            false, a, false, b, (short)0, acc[t], false, false);
      }
    }
  }
  for (int kt = 0; kt < ktiles; ++kt) {
    v16h a = load_a_f16(A2, arow, K, kt, lane);
#pragma unroll
    for (int t = 0; t < NT; ++t) {
      v16h b = load_b(B2, (nt0 + t) * 16 + colbase, K, kt, lane);
      acc[t] = __builtin_amdgcn_wmma_f32_16x16x32_f16(
          false, a, false, b, (short)0, acc[t], false, false);
    }
  }
  // C/D layout: VGPR r -> M = r + 8*(lane>>4); N = lane&15 within the tile.
  int mbase = (lane >> 4) * 8;
#pragma unroll
  for (int t = 0; t < NT; ++t) {
    int col = (nt0 + t) * 16 + colbase;
    float bv = bias[col];
#pragma unroll
    for (int r = 0; r < 8; ++r) {
      float v = fmaxf(acc[t][r] + bv, 0.0f);
      Out[(size_t)(row0 + mbase + r) * Ncol + col] = v;
    }
  }
}

// ---------------- reductions / quant ---------------------------------------

__global__ void minmax_partial(const float* __restrict__ a, long long na,
                               const float* __restrict__ b, long long nb,
                               float* __restrict__ pmin, float* __restrict__ pmax) {
  __shared__ float smin[256], smax[256];
  int tid = threadIdx.x;
  float lo = 3.402823466e38f, hi = -3.402823466e38f;
  long long total = na + nb;
  for (long long i = (long long)blockIdx.x * blockDim.x + tid; i < total;
       i += (long long)gridDim.x * blockDim.x) {
    float v = (i < na) ? a[i] : b[i - na];
    lo = fminf(lo, v);
    hi = fmaxf(hi, v);
  }
  smin[tid] = lo; smax[tid] = hi;
  __syncthreads();
  for (int s = 128; s > 0; s >>= 1) {
    if (tid < s) {
      smin[tid] = fminf(smin[tid], smin[tid + s]);
      smax[tid] = fmaxf(smax[tid], smax[tid + s]);
    }
    __syncthreads();
  }
  if (tid == 0) { pmin[blockIdx.x] = smin[0]; pmax[blockIdx.x] = smax[0]; }
}

__global__ void minmax_final(const float* __restrict__ pmin,
                             const float* __restrict__ pmax,
                             const int* __restrict__ bw,
                             float* __restrict__ prm) {
  __shared__ float smin[256], smax[256];
  int tid = threadIdx.x;
  smin[tid] = pmin[tid]; smax[tid] = pmax[tid];
  __syncthreads();
  for (int s = 128; s > 0; s >>= 1) {
    if (tid < s) {
      smin[tid] = fminf(smin[tid], smin[tid + s]);
      smax[tid] = fmaxf(smax[tid], smax[tid + s]);
    }
    __syncthreads();
  }
  if (tid == 0) {
    float lo = smin[0], hi = smax[0];
    float lv = (float)((1 << bw[0]) - 1);
    float scale = (hi - lo) / lv;
    prm[0] = scale;
    prm[1] = 1.0f / scale;
    prm[2] = rintf(-lo / scale);   // zero point (round-nearest-even)
    prm[3] = lv;
  }
}

// flat fake-quant: f32 -> f16 (row stride unchanged)
__global__ void quant_to_f16(const float* __restrict__ x, _Float16* __restrict__ o,
                             long long n, const float* __restrict__ prm) {
  long long i = (long long)blockIdx.x * blockDim.x + threadIdx.x;
  if (i >= n) return;
  float scale = prm[0], inv = prm[1], zp = prm[2], lv = prm[3];
  float q = fminf(fmaxf(rintf(x[i] * inv + zp), 0.0f), lv);
  o[i] = (_Float16)((q - zp) * scale);
}

// fake-quant with K -> Kpad zero padding: x [rows,K] f32 -> o [rows,Kpad] f16
__global__ void quant_to_f16_pad(const float* __restrict__ x,
                                 _Float16* __restrict__ o,
                                 long long rows, int K, int Kpad,
                                 const float* __restrict__ prm) {
  long long i = (long long)blockIdx.x * blockDim.x + threadIdx.x;
  if (i >= rows * Kpad) return;
  long long row = i / Kpad;
  int col = (int)(i % Kpad);
  if (col >= K) { o[i] = (_Float16)0.0f; return; }
  float scale = prm[0], inv = prm[1], zp = prm[2], lv = prm[3];
  float v = x[row * K + col];
  float q = fminf(fmaxf(rintf(v * inv + zp), 0.0f), lv);
  o[i] = (_Float16)((q - zp) * scale);
}

__global__ void quant_to_f32(const float* __restrict__ x, float* __restrict__ o,
                             long long n, const float* __restrict__ prm) {
  long long i = (long long)blockIdx.x * blockDim.x + threadIdx.x;
  if (i >= n) return;
  float scale = prm[0], inv = prm[1], zp = prm[2], lv = prm[3];
  float q = fminf(fmaxf(rintf(x[i] * inv + zp), 0.0f), lv);
  o[i] = (q - zp) * scale;
}

__global__ void fill_zero(float* __restrict__ p, long long n) {
  long long i = (long long)blockIdx.x * blockDim.x + threadIdx.x;
  if (i < n) p[i] = 0.0f;
}

// W [K,N] f32 -> Wt [N,Kpad] f16 (transposed + zero-padded along K)
__global__ void w_to_f16t(const float* __restrict__ W, _Float16* __restrict__ Wt,
                          int K, int Kpad, int Ncol) {
  long long i = (long long)blockIdx.x * blockDim.x + threadIdx.x;
  if (i >= (long long)Kpad * Ncol) return;
  int n = (int)(i / Kpad), k = (int)(i % Kpad);
  Wt[i] = (k < K) ? (_Float16)W[(size_t)k * Ncol + n] : (_Float16)0.0f;
}

// ---------------- edge message + scatter-add --------------------------------
// agg[dst] += w_e * qx[src]; rows of qx/agg have stride Cs (>= C, zero padded)
__global__ void scatter_edges(const int* __restrict__ ei,
                              const float* __restrict__ qe,
                              const _Float16* __restrict__ qx,
                              float* __restrict__ agg, int E, int C, int Cs) {
  int e = blockIdx.x * blockDim.x + threadIdx.x;
  if (e >= E) return;
  int src = ei[e], dst = ei[E + e];
  float w = qe[e];
  const _Float16* xs = qx + (size_t)src * Cs;
  float* ad = agg + (size_t)dst * Cs;
  if ((C & 3) == 0) {
    for (int c = 0; c < C; c += 4) {
      v4h h = *(const v4h*)(xs + c);
      atomicAdd(ad + c + 0, w * (float)h[0]);
      atomicAdd(ad + c + 1, w * (float)h[1]);
      atomicAdd(ad + c + 2, w * (float)h[2]);
      atomicAdd(ad + c + 3, w * (float)h[3]);
    }
  } else {
    for (int c = 0; c < C; ++c) atomicAdd(ad + c, w * (float)xs[c]);
  }
}

// ---------------- pooling ---------------------------------------------------

__global__ void pool_sum(const float* __restrict__ x, const int* __restrict__ batch,
                         float* __restrict__ sums, float* __restrict__ cnts,
                         int N, int C) {
  int n = blockIdx.x * blockDim.x + threadIdx.x;
  if (n >= N) return;
  int g = batch[n];
  atomicAdd(&cnts[g], 1.0f);
  const float* xr = x + (size_t)n * C;
  float* sr = sums + (size_t)g * C;
  for (int c = 0; c < C; c += 4) {
    float4 v = *(const float4*)(xr + c);
    atomicAdd(sr + c + 0, v.x);
    atomicAdd(sr + c + 1, v.y);
    atomicAdd(sr + c + 2, v.z);
    atomicAdd(sr + c + 3, v.w);
  }
}

__global__ void pool_div(float* __restrict__ sums, const float* __restrict__ cnts,
                         int G, int C) {
  int i = blockIdx.x * blockDim.x + threadIdx.x;
  if (i >= G * C) return;
  sums[i] /= fmaxf(cnts[i / C], 1.0f);
}

// ---------------- 64 -> 1 head ----------------------------------------------

__global__ void final_dot(const _Float16* __restrict__ xq,
                          const float* __restrict__ Wo,
                          const float* __restrict__ bo,
                          float* __restrict__ out, int G, int K) {
  int g = blockIdx.x * blockDim.x + threadIdx.x;
  if (g >= G) return;
  float s = bo[0];
  for (int k = 0; k < K; ++k) s += (float)xq[(size_t)g * K + k] * Wo[k];
  out[g] = s;
}

// ---------------- orchestration ---------------------------------------------

extern "C" void kernel_launch(void* const* d_in, const int* in_sizes, int n_in,
                              void* d_out, int out_size, void* d_ws, size_t ws_size,
                              hipStream_t stream) {
  (void)in_sizes; (void)n_in; (void)out_size; (void)ws_size;
  const float* x_in  = (const float*)d_in[0];
  const int*   ei    = (const int*)d_in[1];
  const float* ea_in = (const float*)d_in[2];
  const int*   batch = (const int*)d_in[3];
  const int*   bw    = (const int*)d_in[4];

  // ---- workspace partition ----
  char* ws = (char*)d_ws;
  size_t off = 0;
  auto take = [&](size_t bytes) -> void* {
    void* p = ws + off;
    off += (bytes + 255) & ~(size_t)255;
    return p;
  };
  float*    xA   = (float*)take((size_t)N_NODES * 512 * 4);
  float*    xB   = (float*)take((size_t)N_NODES * 512 * 4);
  _Float16* qx   = (_Float16*)take((size_t)N_NODES * 512 * 2);
  float*    agg  = (float*)take((size_t)N_NODES * 512 * 4);
  float*    qe   = (float*)take((size_t)N_EDGES * 4);
  _Float16* WtA  = (_Float16*)take((size_t)512 * 512 * 2);
  _Float16* WtB  = (_Float16*)take((size_t)512 * 512 * 2);
  float*    sums = (float*)take((size_t)N_GRAPHS * 256 * 4);
  float*    mbuf = (float*)take((size_t)N_GRAPHS * 256 * 4);
  float*    cnts = (float*)take((size_t)N_GRAPHS * 4);
  float*    pmin = (float*)take(256 * 4);
  float*    pmax = (float*)take(256 * 4);
  float*    prm  = (float*)take(4 * 4);

  const int chans[8] = {5, 32, 128, 256, 512, 512, 256, 256};
  const float* cur_x = x_in;
  const float* cur_e = ea_in;
  float* outbuf = xA;

  // ---- 7 GraphConv layers ----
  for (int l = 0; l < 7; ++l) {
    int cin = chans[l], cout = chans[l + 1];
    int cinP = (l == 0) ? 32 : cin;      // layer 0: zero-pad K 5 -> 32
    const float* Wrel  = (const float*)d_in[5 + 3 * l];
    const float* brel  = (const float*)d_in[6 + 3 * l];
    const float* Wroot = (const float*)d_in[7 + 3 * l];
    long long nx  = (long long)N_NODES * cin;   // logical elements
    long long nxp = (long long)N_NODES * cinP;  // padded elements

    minmax_partial<<<256, 256, 0, stream>>>(cur_x, nx, cur_e, N_EDGES, pmin, pmax);
    minmax_final<<<1, 256, 0, stream>>>(pmin, pmax, bw, prm);
    if (l == 0) {
      quant_to_f16_pad<<<(unsigned)CDIV(nxp, 256), 256, 0, stream>>>(
          cur_x, qx, N_NODES, cin, cinP, prm);
    } else {
      quant_to_f16<<<(unsigned)CDIV(nx, 256), 256, 0, stream>>>(cur_x, qx, nx, prm);
    }
    quant_to_f32<<<CDIV(N_EDGES, 256), 256, 0, stream>>>(cur_e, qe, N_EDGES, prm);
    cur_e = qe;

    fill_zero<<<(unsigned)CDIV(nxp, 256), 256, 0, stream>>>(agg, nxp);
    scatter_edges<<<CDIV(N_EDGES, 256), 256, 0, stream>>>(ei, qe, qx, agg,
                                                          N_EDGES, cin, cinP);
    w_to_f16t<<<(unsigned)CDIV((long long)cinP * cout, 256), 256, 0, stream>>>(
        Wrel, WtA, cin, cinP, cout);
    w_to_f16t<<<(unsigned)CDIV((long long)cinP * cout, 256), 256, 0, stream>>>(
        Wroot, WtB, cin, cinP, cout);

    // cout=32 -> NT=2; cout in {128,256,512} -> NT=4. K (=cinP) % 32 == 0 always.
    if (cout == 32) {
      gemm_kernel<2, true><<<dim3(CDIV(N_NODES, 128), cout / 32), 256, 0, stream>>>(
          agg, WtA, qx, WtB, brel, outbuf, N_NODES, cinP, cout);
    } else {
      gemm_kernel<4, true><<<dim3(CDIV(N_NODES, 128), cout / 64), 256, 0, stream>>>(
          agg, WtA, qx, WtB, brel, outbuf, N_NODES, cinP, cout);
    }

    cur_x = outbuf;
    outbuf = (outbuf == xA) ? xB : xA;
  }

  // ---- global mean pool ----
  fill_zero<<<CDIV(N_GRAPHS * 256, 256), 256, 0, stream>>>(sums,
                                                           (long long)N_GRAPHS * 256);
  fill_zero<<<CDIV(N_GRAPHS, 256), 256, 0, stream>>>(cnts, N_GRAPHS);
  pool_sum<<<CDIV(N_NODES, 256), 256, 0, stream>>>(cur_x, batch, sums, cnts,
                                                   N_NODES, 256);
  pool_div<<<CDIV(N_GRAPHS * 256, 256), 256, 0, stream>>>(sums, cnts, N_GRAPHS, 256);

  // ---- MLP ----
  const int mch[4] = {256, 256, 128, 64};
  float* mx = sums;
  float* mn = mbuf;
  for (int j = 0; j < 3; ++j) {
    const float* W = (const float*)d_in[26 + 2 * j];
    const float* b = (const float*)d_in[27 + 2 * j];
    long long nx = (long long)N_GRAPHS * mch[j];
    minmax_partial<<<256, 256, 0, stream>>>(mx, nx, qe, N_EDGES, pmin, pmax);
    minmax_final<<<1, 256, 0, stream>>>(pmin, pmax, bw, prm);
    quant_to_f16<<<(unsigned)CDIV(nx, 256), 256, 0, stream>>>(mx, qx, nx, prm);
    w_to_f16t<<<CDIV(mch[j] * mch[j + 1], 256), 256, 0, stream>>>(
        W, WtA, mch[j], mch[j], mch[j + 1]);
    // cout: 256/128/64 all divisible by 64; K: 256/256/128 all divisible by 32.
    gemm_kernel<4, false><<<dim3(CDIV(N_GRAPHS, 128), mch[j + 1] / 64),
                            256, 0, stream>>>(
        nullptr, nullptr, qx, WtA, b, mn, N_GRAPHS, mch[j], mch[j + 1]);
    float* t = mx; mx = mn; mn = t;
  }

  // ---- output head ----
  const float* Wo = (const float*)d_in[32];
  const float* bo = (const float*)d_in[33];
  long long nx = (long long)N_GRAPHS * 64;
  minmax_partial<<<256, 256, 0, stream>>>(mx, nx, qe, N_EDGES, pmin, pmax);
  minmax_final<<<1, 256, 0, stream>>>(pmin, pmax, bw, prm);
  quant_to_f16<<<(unsigned)CDIV(nx, 256), 256, 0, stream>>>(mx, qx, nx, prm);
  final_dot<<<CDIV(N_GRAPHS, 256), 256, 0, stream>>>(qx, Wo, bo, (float*)d_out,
                                                     N_GRAPHS, 64);
}
